// EdgeDecoder_42588895707650
// MI455X (gfx1250) — matrix operand order repaired and checked by reading, not compiled, not run on hardware
//
#include <hip/hip_runtime.h>
#include <hip/hip_fp16.h>

typedef __attribute__((ext_vector_type(16))) _Float16 v16h;
typedef __attribute__((ext_vector_type(8)))  float    v8f;
typedef __attribute__((ext_vector_type(4)))  float    v4f;
typedef __attribute__((ext_vector_type(4)))  int      v4i;

typedef __attribute__((address_space(1))) v4i as1_v4i;   // global
typedef __attribute__((address_space(3))) v4i as3_v4i;   // LDS

#define DCH 128   // channels

// ---------------------------------------------------------------------------
// Kernel 1: pack W1 (fp32 [256,128]) into f16 B-fragment order.
// Flat pack index p = (((c*8 + t)*32) + lane)*16 + j
//   c = K-chunk (0..7, 32 K each), t = N-tile (0..7), lane = 0..31, j = 0..15
//   lane: n = t*16 + (lane&15), khalf = lane>>4
//   j:    vgpr v = j>>1, pos = j&1  ->  k = c*32 + khalf*16 + 2*v + pos
// ---------------------------------------------------------------------------
__global__ void pack_w1_kernel(const float* __restrict__ W1,
                               _Float16* __restrict__ W1h) {
  int p = blockIdx.x * blockDim.x + threadIdx.x;
  if (p >= 256 * DCH) return;
  int j    = p & 15;
  int lane = (p >> 4) & 31;
  int t    = (p >> 9) & 7;
  int c    = p >> 12;
  int khalf = lane >> 4;
  int n     = t * 16 + (lane & 15);
  int v     = j >> 1;
  int pos   = j & 1;
  int k     = c * 32 + khalf * 16 + 2 * v + pos;
  W1h[p] = (_Float16)W1[k * DCH + n];
}

// ---------------------------------------------------------------------------
// Kernel 2: one wave per 16-edge tile. W1h staged block-wide in LDS (64 KB)
// via async global->LDS; 8 Kx32 chunks x 8 N-tiles of
// v_wmma_f32_16x16x32_f16 fed from ds_load; fused MLP-head epilogue.
// ---------------------------------------------------------------------------
__global__ __launch_bounds__(256)
void edge_decoder_kernel(const float* __restrict__ src_emb,
                         const float* __restrict__ dst_emb,
                         const long long* __restrict__ eidx,   // [2, E] int64
                         const _Float16* __restrict__ W1h,     // packed 64KB
                         const float* __restrict__ b1,         // [128]
                         const float* __restrict__ W2,         // [128]
                         const float* __restrict__ b2,         // [1]
                         float* __restrict__ out,              // [E]
                         int nEdges) {
  __shared__ _Float16 sW[256 * DCH];   // 64 KB staged weights

  const int tid  = threadIdx.x;
  const int lane = tid & 31;
  const int wave = tid >> 5;

  // ---- cooperative stage of W1h into LDS: 4096 x 16B, 16 per thread ----
#if __has_builtin(__builtin_amdgcn_global_load_async_to_lds_b128)
  {
#pragma unroll
    for (int i = 0; i < 16; ++i) {
      const int e16 = tid + i * 256;   // which 16-byte quantum
      __builtin_amdgcn_global_load_async_to_lds_b128(
          (as1_v4i*)(W1h + e16 * 8),
          (as3_v4i*)(sW + e16 * 8),
          /*offset=*/0, /*cpol=*/0);
    }
#if __has_builtin(__builtin_amdgcn_s_wait_asynccnt)
    __builtin_amdgcn_s_wait_asynccnt(0);
#else
    asm volatile("s_wait_asynccnt 0x0" ::: "memory");
#endif
  }
#else
  {
    const v4i* __restrict__ g = (const v4i*)W1h;
    v4i* s = (v4i*)sW;
#pragma unroll
    for (int i = 0; i < 16; ++i) {
      const int e16 = tid + i * 256;
      s[e16] = g[e16];
    }
  }
#endif
  __syncthreads();

  const int tile = blockIdx.x * 8 + wave;
  const int e0   = tile * 16;
  if (e0 >= nEdges) return;

  const int m  = lane & 15;   // edge row within tile (A-matrix M)
  const int hi = lane >> 4;   // K-half selector for A fragment

  int em = e0 + m; if (em >= nEdges) em = nEdges - 1;
  const long long si = eidx[em];
  const long long di = eidx[(long long)nEdges + em];
  const float* __restrict__ srow = src_emb + si * DCH;
  const float* __restrict__ drow = dst_emb + di * DCH;

  v8f acc[8] = {};   // 8 N-tiles of 16x16 f32 accumulators

#pragma unroll
  for (int c = 0; c < 8; ++c) {
    const float* row = (c < 4) ? srow : drow;
    const int base = (c & 3) * 32;
    // A as loaded here (k contiguous, same as W1 pack): lane hi=0 -> K {base+0..15}
    //                           lane hi=1 -> K {base+16..31}
    const float* r0 = row + base + hi * 16;       // 8 contiguous f32
    const float* r1 = row + base + hi * 16 + 8;   // 8 contiguous f32
    v4f f0 = *(const v4f*)(r0);
    v4f f1 = *(const v4f*)(r0 + 4);
    v4f f2 = *(const v4f*)(r1);
    v4f f3 = *(const v4f*)(r1 + 4);
    v16h a;
    a[0] = (_Float16)f0[0];  a[1] = (_Float16)f0[1];
    a[2] = (_Float16)f0[2];  a[3] = (_Float16)f0[3];
    a[4] = (_Float16)f1[0];  a[5] = (_Float16)f1[1];
    a[6] = (_Float16)f1[2];  a[7] = (_Float16)f1[3];
    a[8]  = (_Float16)f2[0]; a[9]  = (_Float16)f2[1];
    a[10] = (_Float16)f2[2]; a[11] = (_Float16)f2[3];
    a[12] = (_Float16)f3[0]; a[13] = (_Float16)f3[1];
    a[14] = (_Float16)f3[2]; a[15] = (_Float16)f3[3];

#pragma unroll
    for (int t = 0; t < 8; ++t) {
      v16h b = *(const v16h*)(sW + (((c * 8 + t) * 32 + lane) << 4));
      acc[t] = __builtin_amdgcn_wmma_f32_16x16x32_f16(
          /*neg_a=*/false, a, /*neg_b=*/false, b,
          /*c_mod=*/(short)0, acc[t], /*reuse_a=*/false, /*reuse_b=*/false);
    }
  }

  // Epilogue: h = relu(acc + b1[n]); partial[r] = sum_n h * W2[n]
  // C layout: lane owns column n = t*16 + m; VGPR r -> edge row (hi*8 + r).
  const float bb2 = b2[0];
  float partial[8] = {};
#pragma unroll
  for (int t = 0; t < 8; ++t) {
    const int nt  = t * 16 + m;
    const float bias = b1[nt];
    const float w2   = W2[nt];
#pragma unroll
    for (int r = 0; r < 8; ++r) {
      float h = acc[t][r] + bias;
      h = h > 0.0f ? h : 0.0f;
      partial[r] = fmaf(h, w2, partial[r]);
    }
  }
  // reduce across the 16 lanes of each half-wave (columns 0..15)
#pragma unroll
  for (int r = 0; r < 8; ++r) {
    float s = partial[r];
    s += __shfl_xor(s, 1, 32);
    s += __shfl_xor(s, 2, 32);
    s += __shfl_xor(s, 4, 32);
    s += __shfl_xor(s, 8, 32);
    partial[r] = s;
  }
  if (m == 0) {
#pragma unroll
    for (int r = 0; r < 8; ++r) {
      const int e = e0 + hi * 8 + r;
      if (e < nEdges) {
        const float x = partial[r] + bb2;
        out[e] = 1.0f / (1.0f + __expf(-x));
      }
    }
  }
}

// ---------------------------------------------------------------------------
extern "C" void kernel_launch(void* const* d_in, const int* in_sizes, int n_in,
                              void* d_out, int out_size, void* d_ws, size_t ws_size,
                              hipStream_t stream) {
  const float*     src_emb = (const float*)d_in[0];
  const float*     dst_emb = (const float*)d_in[1];
  const long long* eidx    = (const long long*)d_in[2];   // int64 [2, E]
  const float*     W1      = (const float*)d_in[3];       // [256, 128]
  const float*     b1      = (const float*)d_in[4];       // [128]
  const float*     W2      = (const float*)d_in[5];       // [128, 1]
  const float*     b2      = (const float*)d_in[6];       // [1]
  float*           out     = (float*)d_out;

  const int nEdges = in_sizes[2] / 2;

  _Float16* W1h = (_Float16*)d_ws;  // 256*128*2 = 64 KB scratch

  // pack W1 into f16 B-fragment order (deterministic, re-run every call)
  {
    const int total = 256 * DCH;
    pack_w1_kernel<<<(total + 255) / 256, 256, 0, stream>>>(W1, W1h);
  }

  // main fused gather + WMMA GEMM + MLP-head kernel
  {
    const int tiles  = (nEdges + 15) / 16;       // 16 edges per wave
    const int blocks = (tiles + 7) / 8;          // 8 waves per block
    edge_decoder_kernel<<<blocks, 256, 0, stream>>>(
        src_emb, dst_emb, eidx, W1h, b1, W2, b2, out, nEdges);
  }
}
